// DualRangeDistillGNN_57853209477541
// MI455X (gfx1250) — compile-verified, matrix operand order checked by reading, not run on hardware
//
#include <hip/hip_runtime.h>
#include <hip/hip_bf16.h>

// ---------------------------------------------------------------------------
// DualRangeDistillGNN on gfx1250 (MI455X, wave32, WMMA)
//
//   h = MLP(concat(aa_emb[aa], atom_emb[atom]))          (WMMA, 2x 32x32 GEMM)
//   for l in 0..1:
//     aggr_{s,l} = segment_sum(relu(h[src] + ea@eW + eb) * mask)   (edge kernel)
//     h_s/h_l    = relu(z@W1+b1)@W2+b2                   (WMMA, 4x 32x32 GEMM)
//     h          = relu(LN(h + h_s + h_l))               (shfl_xor reductions)
//   out = head MLP (32->16->8->8)
//
// All 32x32 weights are pre-swizzled once into per-lane f16 WMMA B fragments;
// LDS staging tiles are f16 so A fragments are two ds_load_b128 per lane.
// ---------------------------------------------------------------------------

typedef __attribute__((ext_vector_type(16))) _Float16 v16h;
typedef __attribute__((ext_vector_type(8)))  _Float16 v8h;
typedef __attribute__((ext_vector_type(8)))  float    v8f;

#define WPB 4   // waves per block for node kernels (4 * 16 = 64 nodes / block)

// --------------- one-time weight pre-swizzle into B fragments --------------
// Packs a 32x32 f32 row-major [k][col] matrix into per-lane f16 fragments:
//   out[lane*32 + i], i<16 -> b0 (cols 0..15), i>=16 -> b1 (cols 16..31)
// Fragment layout per ISA 7.12.2 (16-bit B 32x16, lane = column):
//   lane<16: col=lane,   K = {kh..kh+7, 16+kh..23+kh} with kh=0
//   lane>=16: col=lane-16, kh=8
__global__ void gnn_pack_w(const float* __restrict__ W, _Float16* __restrict__ out)
{
  for (int idx = threadIdx.x; idx < 1024; idx += blockDim.x) {
    const int lane = idx >> 5;
    const int i    = idx & 31;
    const int r    = lane & 15;
    const int kh   = (lane & 16) ? 8 : 0;
    const int ii   = i & 15;
    const int k    = (ii < 8) ? (kh + ii) : (16 + kh + (ii - 8));
    const int col  = (i < 16) ? r : (16 + r);
    out[lane * 32 + i] = (_Float16)W[k * 32 + col];
  }
}

// One wave computes D[16x32] = relu?(A[16x32] @ W + bias) entirely via WMMA.
// aS: LDS f16 row-major [16][32] (A tile).  Wp: packed per-lane B fragments.
// dS: LDS f16 row-major [16][32] output.  Bias/relu applied in f32.
__device__ __forceinline__ void wave_gemm_16x32x32(
    const _Float16* __restrict__ aS, const _Float16* __restrict__ Wp,
    const float* __restrict__ bias, _Float16* __restrict__ dS,
    bool do_relu, int lane)
{
  const int r  = lane & 15;                // row (A) / column (B)
  const int kh = (lane & 16) ? 8 : 0;      // K half selector
  // A fragment: two contiguous 8-half runs -> 2x ds_load_b128
  const _Float16* aRow = aS + r * 32;
  const v8h a_lo = *(const v8h*)(aRow + kh);
  const v8h a_hi = *(const v8h*)(aRow + 16 + kh);
  const v16h a = __builtin_shufflevector(a_lo, a_hi,
      0, 1, 2, 3, 4, 5, 6, 7, 8, 9, 10, 11, 12, 13, 14, 15);
  // B fragments: 64B contiguous per lane -> 4x global_load_b128 (L1-resident)
  const _Float16* wl = Wp + lane * 32;
  const v16h b0 = *(const v16h*)(wl);
  const v16h b1 = *(const v16h*)(wl + 16);

  v8f c0 = {}, c1 = {};
  c0 = __builtin_amdgcn_wmma_f32_16x16x32_f16(false, a, false, b0, (short)0, c0, false, false);
  c1 = __builtin_amdgcn_wmma_f32_16x16x32_f16(false, a, false, b1, (short)0, c1, false, false);

  // C/D layout: c[j] = D(row = j + (lane>=16 ? 8 : 0), col = lane&15 [+16 for c1])
  const int rd = (lane & 16) ? 8 : 0;
  const float bc0 = bias[r], bc1 = bias[16 + r];
#pragma unroll
  for (int j = 0; j < 8; ++j) {
    float v0 = c0[j] + bc0;
    float v1 = c1[j] + bc1;
    if (do_relu) { v0 = fmaxf(v0, 0.f); v1 = fmaxf(v1, 0.f); }
    dS[(rd + j) * 32 + r]      = (_Float16)v0;
    dS[(rd + j) * 32 + 16 + r] = (_Float16)v1;
  }
}

// ----------------------------- embedding + proj ----------------------------
__global__ void gnn_embed_proj(const int* __restrict__ aa_idx,
                               const int* __restrict__ atom_idx,
                               const float* __restrict__ aa_emb,
                               const float* __restrict__ atom_emb,
                               const _Float16* __restrict__ W1p, const float* __restrict__ b1,
                               const _Float16* __restrict__ W2p, const float* __restrict__ b2,
                               float* __restrict__ h, int N)
{
  __shared__ _Float16 sA[WPB][16 * 32];
  __shared__ _Float16 sT[WPB][16 * 32];
  const int wave = threadIdx.x >> 5, lane = threadIdx.x & 31;
  const int base = (blockIdx.x * WPB + wave) * 16;
  const bool act = base < N;             // wave-uniform: N % 16 == 0
  _Float16* aS = sA[wave];
  _Float16* tS = sT[wave];
  if (act) {
#pragma unroll
    for (int r = 0; r < 16; ++r) {
      const int n = base + r;
      const float v = (lane < 16)
          ? aa_emb[aa_idx[n] * 16 + lane]
          : atom_emb[atom_idx[n] * 16 + (lane - 16)];
      aS[r * 32 + lane] = (_Float16)v;
    }
  }
  __syncthreads();
  if (act) wave_gemm_16x32x32(aS, W1p, b1, tS, true, lane);
  __syncthreads();
  if (act) wave_gemm_16x32x32(tS, W2p, b2, aS, false, lane);
  __syncthreads();
  if (act) {
#pragma unroll
    for (int r = 0; r < 16; ++r)
      h[(size_t)(base + r) * 32 + lane] = (float)aS[r * 32 + lane];
  }
}

// ----------------------------- zero accumulators ---------------------------
__global__ void gnn_zero2(float* __restrict__ a, float* __restrict__ b, int n)
{
  const int i = blockIdx.x * blockDim.x + threadIdx.x;
  if (i < n) { a[i] = 0.f; b[i] = 0.f; }
}

// ------------------------- fused short+long edge pass ----------------------
// One wave per edge, lane = feature. Messages for both branches from one
// h[src] gather; masked scatter-add via global f32 atomics (aggr is L2-resident).
__global__ void gnn_edges(const float* __restrict__ h,
                          const int* __restrict__ src, const int* __restrict__ dst,
                          const float* __restrict__ edge_attr,
                          const int* __restrict__ atom_idx,
                          const float* __restrict__ eWs, const float* __restrict__ ebs,
                          const float* __restrict__ eWl, const float* __restrict__ ebl,
                          float* __restrict__ aggr_s, float* __restrict__ aggr_l, int E)
{
  const int e    = (int)((blockIdx.x * blockDim.x + threadIdx.x) >> 5);
  const int lane = threadIdx.x & 31;
  if (e >= E) return;
  const int s = src[e], d = dst[e];
  const float4 a4 = ((const float4*)edge_attr)[e];
  const float dist = a4.x;                     // uniform in [0,30)
  const bool in_s = (dist >= 0.f) && (dist < 10.f);
  const bool in_l = (dist >= 0.f) && (dist <= 25.f) &&
                    (atom_idx[s] == 1) && (atom_idx[d] == 1);
  if (!in_s && !in_l) return;
  const float hs = h[(size_t)s * 32 + lane];
  if (in_s) {
    float m = hs + a4.x * eWs[lane]      + a4.y * eWs[32 + lane]
                 + a4.z * eWs[64 + lane] + a4.w * eWs[96 + lane] + ebs[lane];
    atomicAdd(&aggr_s[(size_t)d * 32 + lane], fmaxf(m, 0.f));
  }
  if (in_l) {
    float m = hs + a4.x * eWl[lane]      + a4.y * eWl[32 + lane]
                 + a4.z * eWl[64 + lane] + a4.w * eWl[96 + lane] + ebl[lane];
    atomicAdd(&aggr_l[(size_t)d * 32 + lane], fmaxf(m, 0.f));
  }
}

// --------------------- node update: 2x GINE MLP + LN + relu ----------------
__global__ void gnn_node_update(float* __restrict__ h,
                                const float* __restrict__ aggr_s,
                                const float* __restrict__ aggr_l,
                                const float* __restrict__ eps_s_arr,
                                const float* __restrict__ eps_l_arr, int layer,
                                const _Float16* __restrict__ W1sp, const float* __restrict__ b1s,
                                const _Float16* __restrict__ W2sp, const float* __restrict__ b2s,
                                const _Float16* __restrict__ W1lp, const float* __restrict__ b1l,
                                const _Float16* __restrict__ W2lp, const float* __restrict__ b2l,
                                const float* __restrict__ ln_g, const float* __restrict__ ln_b,
                                int N)
{
  __shared__ _Float16 sZ[WPB][16 * 32];
  __shared__ _Float16 sT[WPB][16 * 32];
  const int wave = threadIdx.x >> 5, lane = threadIdx.x & 31;
  const int base = (blockIdx.x * WPB + wave) * 16;
  const bool act = base < N;
  _Float16* zS = sZ[wave];
  _Float16* tS = sT[wave];
  const float es = 1.f + eps_s_arr[layer];
  const float el = 1.f + eps_l_arr[layer];

  float hv[16], acc[16];
  if (act) {
#pragma unroll
    for (int r = 0; r < 16; ++r) {
      const size_t o = (size_t)(base + r) * 32 + lane;
      hv[r]  = h[o];
      acc[r] = hv[r];
      zS[r * 32 + lane] = (_Float16)(es * hv[r] + aggr_s[o]);
    }
  }
  __syncthreads();
  if (act) wave_gemm_16x32x32(zS, W1sp, b1s, tS, true, lane);
  __syncthreads();
  if (act) wave_gemm_16x32x32(tS, W2sp, b2s, zS, false, lane);
  __syncthreads();
  if (act) {
#pragma unroll
    for (int r = 0; r < 16; ++r) acc[r] += (float)zS[r * 32 + lane];
#pragma unroll
    for (int r = 0; r < 16; ++r)
      zS[r * 32 + lane] = (_Float16)(el * hv[r] + aggr_l[(size_t)(base + r) * 32 + lane]);
  }
  __syncthreads();
  if (act) wave_gemm_16x32x32(zS, W1lp, b1l, tS, true, lane);
  __syncthreads();
  if (act) wave_gemm_16x32x32(tS, W2lp, b2l, zS, false, lane);
  __syncthreads();
  if (act) {
    const float g  = ln_g[layer * 32 + lane];
    const float bb = ln_b[layer * 32 + lane];
#pragma unroll
    for (int r = 0; r < 16; ++r) {
      const float x = acc[r] + (float)zS[r * 32 + lane];
      float s = x;                                   // row mean across 32 lanes
#pragma unroll
      for (int off = 16; off > 0; off >>= 1) s += __shfl_xor(s, off, 32);
      const float mu = s * (1.f / 32.f);
      float v = (x - mu) * (x - mu);                 // row var across 32 lanes
#pragma unroll
      for (int off = 16; off > 0; off >>= 1) v += __shfl_xor(v, off, 32);
      const float y = (x - mu) * rsqrtf(v * (1.f / 32.f) + 1e-5f) * g + bb;
      h[(size_t)(base + r) * 32 + lane] = fmaxf(y, 0.f);
    }
  }
}

// --------------------------------- head MLP --------------------------------
__global__ void gnn_head(const float* __restrict__ h,
                         const float* __restrict__ W1, const float* __restrict__ b1,
                         const float* __restrict__ W2, const float* __restrict__ b2,
                         const float* __restrict__ W3, const float* __restrict__ b3,
                         float* __restrict__ out, int N)
{
  const int n = blockIdx.x * blockDim.x + threadIdx.x;
  if (n >= N) return;
  float x[32];
#pragma unroll
  for (int k = 0; k < 32; ++k) x[k] = h[(size_t)n * 32 + k];
  float t1[16];
#pragma unroll
  for (int j = 0; j < 16; ++j) {
    float s = b1[j];
#pragma unroll
    for (int k = 0; k < 32; ++k) s += x[k] * W1[k * 16 + j];
    t1[j] = fmaxf(s, 0.f);
  }
  float t2[8];
#pragma unroll
  for (int j = 0; j < 8; ++j) {
    float s = b2[j];
#pragma unroll
    for (int k = 0; k < 16; ++k) s += t1[k] * W2[k * 8 + j];
    t2[j] = fmaxf(s, 0.f);
  }
#pragma unroll
  for (int j = 0; j < 8; ++j) {
    float s = b3[j];
#pragma unroll
    for (int k = 0; k < 8; ++k) s += t2[k] * W3[k * 8 + j];
    out[(size_t)n * 8 + j] = s;
  }
}

// --------------------------------- launcher --------------------------------
extern "C" void kernel_launch(void* const* d_in, const int* in_sizes, int n_in,
                              void* d_out, int out_size, void* d_ws, size_t ws_size,
                              hipStream_t stream)
{
  const int N = in_sizes[0];
  const int E = in_sizes[3] / 4;

  const int*   aa_idx    = (const int*)d_in[0];
  const int*   atom_idx  = (const int*)d_in[1];
  const int*   edge_idx  = (const int*)d_in[2];     // [2][E]: src then dst
  const float* edge_attr = (const float*)d_in[3];
  const float* aa_emb    = (const float*)d_in[4];
  const float* atom_emb  = (const float*)d_in[5];
  const float* proj_W1   = (const float*)d_in[6];
  const float* proj_b1   = (const float*)d_in[7];
  const float* proj_W2   = (const float*)d_in[8];
  const float* proj_b2   = (const float*)d_in[9];
  const float* short_eps = (const float*)d_in[10];
  const float* short_eW  = (const float*)d_in[11];
  const float* short_eb  = (const float*)d_in[12];
  const float* short_W1  = (const float*)d_in[13];
  const float* short_b1  = (const float*)d_in[14];
  const float* short_W2  = (const float*)d_in[15];
  const float* short_b2  = (const float*)d_in[16];
  const float* long_eps  = (const float*)d_in[17];
  const float* long_eW   = (const float*)d_in[18];
  const float* long_eb   = (const float*)d_in[19];
  const float* long_W1   = (const float*)d_in[20];
  const float* long_b1   = (const float*)d_in[21];
  const float* long_W2   = (const float*)d_in[22];
  const float* long_b2   = (const float*)d_in[23];
  const float* ln_g      = (const float*)d_in[24];
  const float* ln_b      = (const float*)d_in[25];
  const float* head_W1   = (const float*)d_in[26];
  const float* head_b1   = (const float*)d_in[27];
  const float* head_W2   = (const float*)d_in[28];
  const float* head_b2   = (const float*)d_in[29];
  const float* head_W3   = (const float*)d_in[30];
  const float* head_b3   = (const float*)d_in[31];

  // Workspace: h, aggr_s, aggr_l (3 * N * 32 f32 = 38.4 MB; L2-resident),
  // then 10 packed 32x32 weight matrices (1024 f16 each, 20 KB).
  float*     h      = (float*)d_ws;
  float*     aggr_s = h + (size_t)N * 32;
  float*     aggr_l = aggr_s + (size_t)N * 32;
  _Float16*  wpack  = (_Float16*)(aggr_l + (size_t)N * 32);
  // pack order: projW1, projW2, {sW1,sW2,lW1,lW2} x layer0, x layer1
  _Float16* p_projW1 = wpack + 0 * 1024;
  _Float16* p_projW2 = wpack + 1 * 1024;
  _Float16* p_sW1[2] = { wpack + 2 * 1024, wpack + 6 * 1024 };
  _Float16* p_sW2[2] = { wpack + 3 * 1024, wpack + 7 * 1024 };
  _Float16* p_lW1[2] = { wpack + 4 * 1024, wpack + 8 * 1024 };
  _Float16* p_lW2[2] = { wpack + 5 * 1024, wpack + 9 * 1024 };

  gnn_pack_w<<<1, 256, 0, stream>>>(proj_W1, p_projW1);
  gnn_pack_w<<<1, 256, 0, stream>>>(proj_W2, p_projW2);
  for (int l = 0; l < 2; ++l) {
    gnn_pack_w<<<1, 256, 0, stream>>>(short_W1 + l * 1024, p_sW1[l]);
    gnn_pack_w<<<1, 256, 0, stream>>>(short_W2 + l * 1024, p_sW2[l]);
    gnn_pack_w<<<1, 256, 0, stream>>>(long_W1  + l * 1024, p_lW1[l]);
    gnn_pack_w<<<1, 256, 0, stream>>>(long_W2  + l * 1024, p_lW2[l]);
  }

  const int nodeBlocks = (N + WPB * 16 - 1) / (WPB * 16);
  const int edgeBlocks = (E + 7) / 8;                 // 8 edges (waves) / 256-thr block
  const int zeroBlocks = (N * 32 + 255) / 256;

  gnn_embed_proj<<<nodeBlocks, WPB * 32, 0, stream>>>(
      aa_idx, atom_idx, aa_emb, atom_emb, p_projW1, proj_b1, p_projW2, proj_b2, h, N);

  for (int l = 0; l < 2; ++l) {
    gnn_zero2<<<zeroBlocks, 256, 0, stream>>>(aggr_s, aggr_l, N * 32);
    gnn_edges<<<edgeBlocks, 256, 0, stream>>>(
        h, edge_idx, edge_idx + E, edge_attr, atom_idx,
        short_eW + l * 128, short_eb + l * 32,
        long_eW + l * 128,  long_eb + l * 32,
        aggr_s, aggr_l, E);
    gnn_node_update<<<nodeBlocks, WPB * 32, 0, stream>>>(
        h, aggr_s, aggr_l, short_eps, long_eps, l,
        p_sW1[l], short_b1 + l * 32, p_sW2[l], short_b2 + l * 32,
        p_lW1[l], long_b1 + l * 32,  p_lW2[l], long_b2 + l * 32,
        ln_g, ln_b, N);
  }

  gnn_head<<<(N + 127) / 128, 128, 0, stream>>>(
      h, head_W1, head_b1, head_W2, head_b2, head_W3, head_b3, (float*)d_out, N);
}
